// AnomalyBlock_49495203119143
// MI455X (gfx1250) — matrix operand order repaired and checked by reading, not compiled, not run on hardware
//
#include <hip/hip_runtime.h>
#include <hip/hip_bf16.h>

// ---------------------------------------------------------------------------
// AnomalyBlock for MI455X (gfx1250).
// All GEMM-like stages run on v_wmma_f32_16x16x32_bf16 with operands stored
// as bf16 in workspace (converted once), so the inner loops are pure
// {global_load_b128 -> v_wmma} with no per-use conversion VALU.
// B=2, S=2048, E=512, H=8, D=64, W=2048.
// ---------------------------------------------------------------------------

#define B_ 2
#define S_ 2048
#define E_ 512
#define H_ 8
#define D_ 64
#define W_ 2048
#define MS_ 4096   // B*S

typedef __attribute__((ext_vector_type(16))) __bf16 v16bf;
typedef __attribute__((ext_vector_type(8)))  __bf16 v8bf;
typedef __attribute__((ext_vector_type(8)))  float  v8f;

// float -> bf16, round-to-nearest-even (bit trick)
__device__ __forceinline__ __bf16 f2bf(float f) {
  union { float f; unsigned u; } in; in.f = f;
  unsigned r = in.u + 0x7FFFu + ((in.u >> 16) & 1u);
  union { unsigned short s; __bf16 b; } out;
  out.s = (unsigned short)(r >> 16);
  return out.b;
}

// A fragment: 16x32 bf16 tile, row-major, leading dim lda (bf16 elements).
// Lanes 0-15: rows 0..15, K=0..15; lanes 16-31: same rows, K=16..31.
__device__ __forceinline__ v16bf load_frag_a_bf16(const __bf16* __restrict__ A,
                                                  int lda, int lane) {
  int row = lane & 15;
  int kb  = (lane >> 4) << 4;   // 0 or 16
  const __bf16* p = A + (size_t)row * lda;
  v8bf lo = *(const v8bf*)(p + kb);        // K kb..kb+7      (16B)
  v8bf hi = *(const v8bf*)(p + kb + 8);    // K kb+8..kb+15   (16B)
  return __builtin_shufflevector(lo, hi, 0,1,2,3,4,5,6,7,8,9,10,11,12,13,14,15);
}

// B fragment: 32x16, element (k,n) = Bt[n*ldk + k] (K contiguous per lane).
// Lanes 0-15: col n, K=0..15; lanes 16-31: col n, K=16..31.  One 32B load.
__device__ __forceinline__ v16bf load_frag_b_nk(const __bf16* __restrict__ Bt,
                                                int ldk, int lane) {
  int n  = lane & 15;
  int kb = (lane >> 4) << 4;    // 0 or 16
  return *(const v16bf*)(Bt + (size_t)n * ldk + kb);
}

// ---------------------------------------------------------------------------
// GEMM: Y[M,N] = A[M,K] @ Wt[N,K]^T + bias[N].  A,Wt bf16; Y -> f32 and/or
// bf16 (either pointer may be null).  One wave computes a 16x64 tile
// (4 accumulators), so each A fragment feeds 4 WMMAs.
// ---------------------------------------------------------------------------
__global__ void gemm_bf16_kernel(const __bf16* __restrict__ A,
                                 const __bf16* __restrict__ Wt,
                                 const float* __restrict__ bias,
                                 float* __restrict__ Yf,
                                 __bf16* __restrict__ Yb,
                                 int M, int N, int K) {
  int lane = threadIdx.x & 31;
  int wave = threadIdx.x >> 5;
  int cgrp = N >> 6;                              // 64-wide column groups
  int tile = blockIdx.x * (blockDim.x >> 5) + wave;
  int total = (M >> 4) * cgrp;
  if (tile >= total) return;                      // wave-uniform
  int mt = tile / cgrp, ng = tile - mt * cgrp;
  int m0 = mt << 4, n0 = ng << 6;

  const __bf16* Ap = A + (size_t)m0 * K;
  v8f acc0 = {}, acc1 = {}, acc2 = {}, acc3 = {};
  for (int k0 = 0; k0 < K; k0 += 32) {
    __builtin_prefetch(Ap + (size_t)(lane & 15) * K + k0 + 128, 0, 1);
    v16bf a = load_frag_a_bf16(Ap + k0, K, lane);
    v16bf b0 = load_frag_b_nk(Wt + (size_t)(n0     ) * K + k0, K, lane);
    v16bf b1 = load_frag_b_nk(Wt + (size_t)(n0 + 16) * K + k0, K, lane);
    v16bf b2 = load_frag_b_nk(Wt + (size_t)(n0 + 32) * K + k0, K, lane);
    v16bf b3 = load_frag_b_nk(Wt + (size_t)(n0 + 48) * K + k0, K, lane);
    acc0 = __builtin_amdgcn_wmma_f32_16x16x32_bf16(false, a, false, b0, (short)0, acc0, false, false);
    acc1 = __builtin_amdgcn_wmma_f32_16x16x32_bf16(false, a, false, b1, (short)0, acc1, false, false);
    acc2 = __builtin_amdgcn_wmma_f32_16x16x32_bf16(false, a, false, b2, (short)0, acc2, false, false);
    acc3 = __builtin_amdgcn_wmma_f32_16x16x32_bf16(false, a, false, b3, (short)0, acc3, false, false);
  }

  int nloc  = lane & 15;
  int mbase = m0 + ((lane >> 4) << 3);
  v8f accs[4] = {acc0, acc1, acc2, acc3};
#pragma unroll
  for (int c = 0; c < 4; ++c) {
    int nn = n0 + 16 * c + nloc;
    float bv = bias ? bias[nn] : 0.0f;
#pragma unroll
    for (int j = 0; j < 8; ++j) {
      float val = accs[c][j] + bv;
      if (Yf) Yf[(size_t)(mbase + j) * N + nn] = val;
      if (Yb) Yb[(size_t)(mbase + j) * N + nn] = f2bf(val);
    }
  }
}

// ---------------------------------------------------------------------------
// Scores: series[b,h,s,t] = (q.k)/8, causal-masked to -inf.
// One wave per 16x64 strip: Q fragments (D=64 -> 2) loaded once, 4 K-subtiles.
// grid = (S/64, S/16, B*H), block = 32.
// ---------------------------------------------------------------------------
__global__ void scores_kernel(const __bf16* __restrict__ q,
                              const __bf16* __restrict__ k,
                              float* __restrict__ series) {
  int lane = threadIdx.x;
  int tg = blockIdx.x, mt = blockIdx.y, bh = blockIdx.z;
  int b = bh >> 3, h = bh & 7;
  int m0 = mt << 4;
  int nloc  = lane & 15;
  int mbase = m0 + ((lane >> 4) << 3);
  float* out = series + (size_t)bh * S_ * S_;

  v16bf qa0, qa1;
  if (tg * 64 <= m0) {   // at least one subtile computes: load Q once
    const __bf16* qp = q + ((size_t)b * S_ + m0) * E_ + h * D_;
    qa0 = load_frag_a_bf16(qp, E_, lane);
    qa1 = load_frag_a_bf16(qp + 32, E_, lane);
  }

#pragma unroll
  for (int c = 0; c < 4; ++c) {
    int t0 = tg * 64 + 16 * c;
    if (t0 > m0) {        // fully masked subtile (wave-uniform)
#pragma unroll
      for (int j = 0; j < 8; ++j)
        out[(size_t)(mbase + j) * S_ + t0 + nloc] = -__builtin_inff();
    } else {
      const __bf16* kp = k + ((size_t)b * S_ + t0) * E_ + h * D_;
      v8f acc = {};
      acc = __builtin_amdgcn_wmma_f32_16x16x32_bf16(false, qa0, false,
              load_frag_b_nk(kp, E_, lane), (short)0, acc, false, false);
      acc = __builtin_amdgcn_wmma_f32_16x16x32_bf16(false, qa1, false,
              load_frag_b_nk(kp + 32, E_, lane), (short)0, acc, false, false);
      const float scale = 0.125f;   // 1/sqrt(64)
#pragma unroll
      for (int j = 0; j < 8; ++j) {
        int mm = mbase + j, nn = t0 + nloc;
        float v = acc[j] * scale;
        if (nn > mm) v = -__builtin_inff();
        out[(size_t)mm * S_ + nn] = v;
      }
    }
  }
}

// ---------------------------------------------------------------------------
// Block reductions
// ---------------------------------------------------------------------------
__device__ __forceinline__ float block_reduce_max(float v, float* red) {
#pragma unroll
  for (int off = 16; off > 0; off >>= 1) v = fmaxf(v, __shfl_xor(v, off, 32));
  int tid = threadIdx.x;
  if ((tid & 31) == 0) red[tid >> 5] = v;
  __syncthreads();
  float r = red[0];
#pragma unroll
  for (int i = 1; i < 8; ++i) r = fmaxf(r, red[i]);
  __syncthreads();
  return r;
}

__device__ __forceinline__ float block_reduce_sum(float v, float* red) {
#pragma unroll
  for (int off = 16; off > 0; off >>= 1) v += __shfl_xor(v, off, 32);
  int tid = threadIdx.x;
  if ((tid & 31) == 0) red[tid >> 5] = v;
  __syncthreads();
  float r = red[0];
#pragma unroll
  for (int i = 1; i < 8; ++i) r += red[i];
  __syncthreads();
  return r;
}

// Row softmax in place; also writes the bf16 copy consumed by the PV GEMM.
__global__ void softmax_kernel(float* __restrict__ series,
                               __bf16* __restrict__ seriesbf) {
  __shared__ float red[8];
  float*  p  = series   + (size_t)blockIdx.x * S_;
  __bf16* pb = seriesbf + (size_t)blockIdx.x * S_;
  int tid = threadIdx.x;
  float m = -__builtin_inff();
  for (int i = tid; i < S_; i += 256) m = fmaxf(m, p[i]);
  m = block_reduce_max(m, red);
  float s = 0.0f;
  for (int i = tid; i < S_; i += 256) s += __expf(p[i] - m);
  s = block_reduce_sum(s, red);
  float inv = 1.0f / s;
  for (int i = tid; i < S_; i += 256) {
    float v = __expf(p[i] - m) * inv;
    p[i]  = v;
    pb[i] = f2bf(v);
  }
}

// Vt[bh, d, t] = v[b, t, h*64 + d]  (bf16): makes PV's B-fragments contiguous.
__global__ void transpose_v_kernel(const __bf16* __restrict__ vbf,
                                   __bf16* __restrict__ Vt) {
  int i = blockIdx.x * 256 + threadIdx.x;   // total B*H*D*S = 2^21
  int t  = i & (S_ - 1);
  int d  = (i >> 11) & (D_ - 1);
  int bh = i >> 17;
  int b = bh >> 3, h = bh & 7;
  Vt[i] = vbf[((size_t)b * S_ + t) * E_ + h * D_ + d];
}

// ---------------------------------------------------------------------------
// PV: attn[b,s,h*64+d] = sum_t series[b,h,s,t] * v[b,t,h*64+d].
// One wave covers a full 16x64 (all of D for one head): series strip is read
// once and feeds 4 WMMAs per k-step.  Causal: k-loop stops at diagonal tile.
// grid = (S/16, B*H), block = 32.  Output bf16 (feeds Wo GEMM).
// ---------------------------------------------------------------------------
__global__ void pv_kernel(const __bf16* __restrict__ seriesbf,
                          const __bf16* __restrict__ Vt,
                          __bf16* __restrict__ attnbf) {
  int lane = threadIdx.x;
  int mt = blockIdx.x, bh = blockIdx.y;
  int b = bh >> 3, h = bh & 7;
  int m0 = mt << 4;
  const __bf16* sp = seriesbf + ((size_t)bh * S_ + m0) * S_;
  const __bf16* vt = Vt + (size_t)bh * D_ * S_;

  v8f acc0 = {}, acc1 = {}, acc2 = {}, acc3 = {};
  int kmax = m0 + 16;     // nonzero columns are t <= m0+15
  for (int k0 = 0; k0 < kmax; k0 += 32) {
    v16bf a = load_frag_a_bf16(sp + k0, S_, lane);
    v16bf b0 = load_frag_b_nk(vt + (size_t) 0 * S_ + k0, S_, lane);
    v16bf b1 = load_frag_b_nk(vt + (size_t)16 * S_ + k0, S_, lane);
    v16bf b2 = load_frag_b_nk(vt + (size_t)32 * S_ + k0, S_, lane);
    v16bf b3 = load_frag_b_nk(vt + (size_t)48 * S_ + k0, S_, lane);
    acc0 = __builtin_amdgcn_wmma_f32_16x16x32_bf16(false, a, false, b0, (short)0, acc0, false, false);
    acc1 = __builtin_amdgcn_wmma_f32_16x16x32_bf16(false, a, false, b1, (short)0, acc1, false, false);
    acc2 = __builtin_amdgcn_wmma_f32_16x16x32_bf16(false, a, false, b2, (short)0, acc2, false, false);
    acc3 = __builtin_amdgcn_wmma_f32_16x16x32_bf16(false, a, false, b3, (short)0, acc3, false, false);
  }

  int nloc  = lane & 15;
  int mbase = m0 + ((lane >> 4) << 3);
  v8f accs[4] = {acc0, acc1, acc2, acc3};
#pragma unroll
  for (int c = 0; c < 4; ++c)
#pragma unroll
    for (int j = 0; j < 8; ++j)
      attnbf[((size_t)b * S_ + mbase + j) * E_ + h * D_ + 16 * c + nloc] =
          f2bf(accs[c][j]);
}

// ---------------------------------------------------------------------------
// y = LayerNorm(a + b) * w + bias.  Optional bf16 copy (for the FFN GEMM).
// ---------------------------------------------------------------------------
__global__ void add_ln_kernel(const float* __restrict__ A,
                              const float* __restrict__ Bres,
                              const float* __restrict__ w,
                              const float* __restrict__ bias,
                              float* __restrict__ Y,
                              __bf16* __restrict__ Yb) {
  __shared__ float red[8];
  int row = blockIdx.x, tid = threadIdx.x;
  const float* pa = A    + (size_t)row * E_;
  const float* pb = Bres + (size_t)row * E_;
  float x0 = pa[tid] + pb[tid];
  float x1 = pa[tid + 256] + pb[tid + 256];
  float mean = block_reduce_sum(x0 + x1, red) * (1.0f / E_);
  float d0 = x0 - mean, d1 = x1 - mean;
  float var = block_reduce_sum(d0 * d0 + d1 * d1, red) * (1.0f / E_);
  float inv = rsqrtf(var + 1e-5f);
  float y0 = d0 * inv * w[tid]       + bias[tid];
  float y1v = d1 * inv * w[tid + 256] + bias[tid + 256];
  float* py = Y + (size_t)row * E_;
  py[tid]       = y0;
  py[tid + 256] = y1v;
  if (Yb) {
    __bf16* pyb = Yb + (size_t)row * E_;
    pyb[tid]       = f2bf(y0);
    pyb[tid + 256] = f2bf(y1v);
  }
}

// prior[b,i,j] = 1/(sqrt(2*pi)*sigma) * exp(-|i-j|^2 / (2*sigma))
__global__ void prior_kernel(const float* __restrict__ sigma,
                             float* __restrict__ prior, int total) {
  int i = blockIdx.x * 256 + threadIdx.x;
  if (i >= total) return;
  int j  = i & (W_ - 1);
  int ii = (i >> 11) & (S_ - 1);
  float s = sigma[i];
  float d = (float)(ii - j);
  const float inv_sqrt2pi = 0.3989422804014327f;
  prior[i] = (inv_sqrt2pi / s) * __expf(-(d * d) / (2.0f * s));
}

// One-time f32 -> bf16 conversion
__global__ void cvt_f32_bf16_kernel(const float* __restrict__ src,
                                    __bf16* __restrict__ dst, int n) {
  int i = blockIdx.x * 256 + threadIdx.x;
  if (i < n) dst[i] = f2bf(src[i]);
}

// ---------------------------------------------------------------------------
// Host launcher
// ---------------------------------------------------------------------------
extern "C" void kernel_launch(void* const* d_in, const int* in_sizes, int n_in,
                              void* d_out, int out_size, void* d_ws, size_t ws_size,
                              hipStream_t stream) {
  const float* x      = (const float*)d_in[0];
  const float* Wq     = (const float*)d_in[1];
  const float* bq     = (const float*)d_in[2];
  const float* Wk     = (const float*)d_in[3];
  const float* bk     = (const float*)d_in[4];
  const float* Wv     = (const float*)d_in[5];
  const float* bv     = (const float*)d_in[6];
  const float* Wsig   = (const float*)d_in[7];
  const float* bsig   = (const float*)d_in[8];
  const float* Wo     = (const float*)d_in[9];
  const float* bo     = (const float*)d_in[10];
  const float* lin1_W = (const float*)d_in[11];
  const float* lin1_b = (const float*)d_in[12];
  const float* ln1_w  = (const float*)d_in[13];
  const float* ln1_b  = (const float*)d_in[14];
  const float* ln2_w  = (const float*)d_in[15];
  const float* ln2_b  = (const float*)d_in[16];

  // ---- workspace carve (256B aligned chunks) ----
  char* wp = (char*)d_ws;
  auto alloc = [&](size_t bytes) -> void* {
    void* r = (void*)wp;
    wp += (bytes + 255) & ~(size_t)255;
    return r;
  };
  const size_t NE = (size_t)MS_ * E_;                 // 2,097,152 elements
  __bf16* xbf      = (__bf16*)alloc(NE * 2);
  __bf16* wqb      = (__bf16*)alloc((size_t)E_ * E_ * 2);
  __bf16* wkb      = (__bf16*)alloc((size_t)E_ * E_ * 2);
  __bf16* wvb      = (__bf16*)alloc((size_t)E_ * E_ * 2);
  __bf16* wob      = (__bf16*)alloc((size_t)E_ * E_ * 2);
  __bf16* l1b      = (__bf16*)alloc((size_t)E_ * E_ * 2);
  __bf16* wsb      = (__bf16*)alloc((size_t)W_ * E_ * 2);
  __bf16* qbf      = (__bf16*)alloc(NE * 2);
  __bf16* kbf      = (__bf16*)alloc(NE * 2);
  __bf16* vbf      = (__bf16*)alloc(NE * 2);
  __bf16* Vt       = (__bf16*)alloc(NE * 2);
  __bf16* attnbf   = (__bf16*)alloc(NE * 2);
  __bf16* y1bf     = (__bf16*)alloc(NE * 2);
  __bf16* seriesbf = (__bf16*)alloc((size_t)B_ * H_ * S_ * S_ * 2);
  float*  sigma    = (float*)alloc((size_t)B_ * S_ * W_ * 4);
  float*  tmp      = (float*)alloc(NE * 4);
  float*  y1       = (float*)alloc(NE * 4);
  float*  tmp2     = (float*)alloc(NE * 4);

  float* out    = (float*)d_out;                       // [B,S,E]
  float* series = out + NE;                            // [B,H,S,S]
  float* prior  = series + (size_t)B_ * H_ * S_ * S_;  // [B,S,W]

  auto cvt = [&](const float* s, __bf16* d, int n) {
    cvt_f32_bf16_kernel<<<(n + 255) / 256, 256, 0, stream>>>(s, d, n);
  };
  auto gemm = [&](const __bf16* A, const __bf16* Wt, const float* bias,
                  float* Yf, __bf16* Yb, int M, int N, int K) {
    int tiles = (M / 16) * (N / 64);
    gemm_bf16_kernel<<<(tiles + 7) / 8, 256, 0, stream>>>(A, Wt, bias, Yf, Yb, M, N, K);
  };

  // One-time bf16 conversions
  cvt(x, xbf, (int)NE);
  cvt(Wq, wqb, E_ * E_);
  cvt(Wk, wkb, E_ * E_);
  cvt(Wv, wvb, E_ * E_);
  cvt(Wo, wob, E_ * E_);
  cvt(lin1_W, l1b, E_ * E_);
  cvt(Wsig, wsb, W_ * E_);

  // Projections (q/k/v only needed as bf16; sigma only as f32)
  gemm(xbf, wqb, bq, nullptr, qbf, MS_, E_, E_);
  gemm(xbf, wkb, bk, nullptr, kbf, MS_, E_, E_);
  gemm(xbf, wvb, bv, nullptr, vbf, MS_, E_, E_);
  gemm(xbf, wsb, bsig, sigma, nullptr, MS_, W_, E_);

  // Attention
  scores_kernel<<<dim3(S_ / 64, S_ / 16, B_ * H_), 32, 0, stream>>>(qbf, kbf, series);
  softmax_kernel<<<B_ * H_ * S_, 256, 0, stream>>>(series, seriesbf);
  transpose_v_kernel<<<(int)(NE / 256), 256, 0, stream>>>(vbf, Vt);
  pv_kernel<<<dim3(S_ / 16, B_ * H_), 32, 0, stream>>>(seriesbf, Vt, attnbf);

  // Output projection + residual/LN + FFN + residual/LN
  gemm(attnbf, wob, bo, tmp, nullptr, MS_, E_, E_);
  add_ln_kernel<<<MS_, 256, 0, stream>>>(tmp, x, ln1_w, ln1_b, y1, y1bf);
  gemm(y1bf, l1b, lin1_b, tmp2, nullptr, MS_, E_, E_);
  add_ln_kernel<<<MS_, 256, 0, stream>>>(tmp2, y1, ln2_w, ln2_b, out, nullptr);

  // Gaussian prior (elementwise)
  int total = B_ * S_ * W_;
  prior_kernel<<<(total + 255) / 256, 256, 0, stream>>>(sigma, prior, total);
}
